// Attention_7834020348232
// MI455X (gfx1250) — matrix operand order repaired and checked by reading, not compiled, not run on hardware
//
#include <hip/hip_runtime.h>
#include <hip/hip_bf16.h>

typedef __attribute__((ext_vector_type(16))) _Float16 v16h;
typedef __attribute__((ext_vector_type(8)))  _Float16 v8h;
typedef __attribute__((ext_vector_type(2)))  _Float16 v2h;
typedef __attribute__((ext_vector_type(8)))  float    v8f;

#define DIMX   512
#define HEADS  8
#define DH     32
#define NTOK   49
#define NPAD   64
#define QKVW   768

// ---- WMMA wrapper: D = A(16x32 f16) * B(32x16 f16) + C(16x16 f32) ----
__device__ __forceinline__ v8f wmma_f16(v16h a, v16h b, v8f c) {
    return __builtin_amdgcn_wmma_f32_16x16x32_f16(
        /*neg_a=*/false, a, /*neg_b=*/false, b,
        /*c_mod=*/(short)0, c, /*reuse_a=*/false, /*reuse_b=*/false);
}

// A-matrix loader (16-bit A 16x32, contiguous k order shared with the B loaders):
// lane L: row M = m0 + (L&15); half = L>>4
//   elems j (0..15) = K = k0 + half*16 + j
__device__ __forceinline__ v16h load_A(const _Float16* base, int stride,
                                       int m0, int k0, int lane) {
    int row = m0 + (lane & 15);
    int hf  = lane >> 4;
    const _Float16* p = base + row * stride + k0 + hf * 16;
    v8h lo = *(const v8h*)(p);
    v8h hi = *(const v8h*)(p + 8);
    return __builtin_shufflevector(lo, hi, 0,1,2,3,4,5,6,7,8,9,10,11,12,13,14,15);
}

// B-matrix loader, "N-major" storage (row n holds K contiguously):
// lane L: col N = n0 + (L&15); elems j = Bsrc[n][k0 + (L>>4)*16 + j]
__device__ __forceinline__ v16h load_B_rows(const _Float16* base, int stride,
                                            int n0, int k0, int lane) {
    int n  = n0 + (lane & 15);
    int hf = lane >> 4;
    return *(const v16h*)(base + n * stride + k0 + hf * 16);
}

// B-matrix loader for K-major storage (stride between K rows): per-element gather.
__device__ __forceinline__ v16h load_B_strided(const _Float16* base, int kstride,
                                               int n0, int k0, int lane) {
    int n  = n0 + (lane & 15);
    int hf = lane >> 4;
    v16h b;
    #pragma unroll
    for (int j = 0; j < 16; j++)
        b[j] = base[(k0 + hf * 16 + j) * kstride + n];
    return b;
}

// ------------- prep kernel: weight transpose->f16, rope tables, wgT -------------
extern "C" __global__ void __launch_bounds__(256)
attn_prep(const float* __restrict__ wqkv, const float* __restrict__ wout,
          const float* __restrict__ rot, const float* __restrict__ wg,
          _Float16* __restrict__ wqkvT, _Float16* __restrict__ woutT,
          float* __restrict__ costab, float* __restrict__ sintab,
          _Float16* __restrict__ wgT) {
    int i = blockIdx.x * 256 + threadIdx.x;
    if (i < QKVW * DIMX) {                 // wqkvT[n][k] = wqkv[k][n]
        int n = i / DIMX, k = i % DIMX;
        wqkvT[i] = (_Float16)wqkv[k * QKVW + n];
    }
    if (i < DIMX * 256) {                  // woutT[n][k] = wout[k][n]
        int n = i / 256, k = i % 256;
        woutT[i] = (_Float16)wout[k * DIMX + n];
    }
    if (i < 16 * DIMX) {                   // wgT[n][k] = wg[k][n], rows 8..15 zero
        int n = i / DIMX, k = i % DIMX;
        wgT[i] = (n < HEADS) ? (_Float16)wg[k * HEADS + n] : (_Float16)0.f;
    }
    if (i < NTOK * DH) {
        float r = rot[i];
        costab[i] = cosf(r);
        sintab[i] = sinf(r);
    }
}

// ---------------- main kernel: one workgroup per 49-token window ----------------
extern "C" __global__ void __launch_bounds__(256)
attn_main(const float* __restrict__ x, const float* __restrict__ gamma,
          const float* __restrict__ bg,
          const _Float16* __restrict__ wqkvT, const _Float16* __restrict__ woutT,
          const float* __restrict__ costab, const float* __restrict__ sintab,
          const _Float16* __restrict__ wgT,
          float* __restrict__ out) {
    extern __shared__ char smem[];
    _Float16* sXn  = (_Float16*)smem;              // 64 x 512 f16 (reused as attn scratch)
    _Float16* sQKV = sXn + NPAD * DIMX;            // 64 x 768 f16 (q|k|v; q-region reused)
    float*    sG   = (float*)(sQKV + NPAD * QKVW); // 64 x 8 f32 gates

    const int tid  = threadIdx.x;
    const int lane = tid & 31;
    const int wid  = tid >> 5;
    const long win = blockIdx.x;
    const float* xw = x   + win * (long)(NTOK * DIMX);
    float*       ow = out + win * (long)(NTOK * DIMX);

    // ---- Phase 1: row L2-norm + gamma*sqrt(d) -> xn f16 (zero-padded to 64 rows) ----
    for (int t = wid; t < NTOK; t += 8) {
        const float4* row4 = (const float4*)(xw + t * DIMX);
        const float4* g4   = (const float4*)gamma;
        float4 v4[4];
        float ss = 0.f;
        #pragma unroll
        for (int i = 0; i < 4; i++) {
            float4 v = row4[lane * 4 + i];
            v4[i] = v;
            ss += v.x * v.x + v.y * v.y + v.z * v.z + v.w * v.w;
        }
        #pragma unroll
        for (int m = 16; m >= 1; m >>= 1) ss += __shfl_xor(ss, m, 32);
        float scale = 22.627416997969522f / fmaxf(sqrtf(ss), 1e-12f); // sqrt(512)/norm
        v8h h0, h1;
        #pragma unroll
        for (int i = 0; i < 4; i++) {
            float4 gv = g4[lane * 4 + i];
            v8h& dst  = (i < 2) ? h0 : h1;
            int  base = (i & 1) * 4;
            dst[base + 0] = (_Float16)(v4[i].x * scale * gv.x);
            dst[base + 1] = (_Float16)(v4[i].y * scale * gv.y);
            dst[base + 2] = (_Float16)(v4[i].z * scale * gv.z);
            dst[base + 3] = (_Float16)(v4[i].w * scale * gv.w);
        }
        *(v8h*)(sXn + t * DIMX + lane * 16)     = h0;
        *(v8h*)(sXn + t * DIMX + lane * 16 + 8) = h1;
    }
    {
        v8h z = {};
        for (int idx = tid; idx < (NPAD - NTOK) * DIMX / 8; idx += 256)
            *(v8h*)(sXn + NTOK * DIMX + idx * 8) = z;
    }
    __syncthreads();

    // ---- Phase 2 (waves 0..3): gates via WMMA: sigmoid(xn @ wgT' + bg) ----
    if (wid < 4) {
        int mi = wid;
        v8f acc = {};
        #pragma unroll
        for (int kk = 0; kk < 16; kk++) {
            v16h a = load_A(sXn, DIMX, mi * 16, kk * 32, lane);
            v16h b = load_B_rows(wgT, DIMX, 0, kk * 32, lane);
            acc = wmma_f16(a, b, acc);
        }
        int h     = lane & 15;
        int rbase = mi * 16 + 8 * (lane >> 4);
        float bgv = (h < HEADS) ? bg[h] : 0.f;
        #pragma unroll
        for (int r = 0; r < 8; r++) {
            int m = rbase + r;
            if (m < NTOK && h < HEADS)
                sG[m * HEADS + h] = 1.f / (1.f + __expf(-(acc[r] + bgv)));
        }
    }

    // ---- Phase 3: qkv = xn(64x512) @ Wqkv via WMMA; A-fragments hoisted ----
    {
        const int mi = wid & 3;          // fixed M-tile per wave
        v16h aF[16];
        #pragma unroll
        for (int kk = 0; kk < 16; kk++)
            aF[kk] = load_A(sXn, DIMX, mi * 16, kk * 32, lane);
        for (int nt = (wid >> 2); nt < 48; nt += 2) {
            v8f acc = {};
            #pragma unroll
            for (int kk = 0; kk < 16; kk++) {
                v16h b = load_B_rows(wqkvT, DIMX, nt * 16, kk * 32, lane);
                acc = wmma_f16(aF[kk], b, acc);
            }
            int col   = nt * 16 + (lane & 15);
            int rbase = mi * 16 + 8 * (lane >> 4);
            #pragma unroll
            for (int r = 0; r < 8; r++)
                sQKV[(rbase + r) * QKVW + col] = (_Float16)acc[r];
        }
    }
    __syncthreads();

    // ---- Phase 4: RoPE on q and k (cols 0..511), pairwise in-place ----
    for (int idx = tid; idx < NTOK * 256; idx += 256) {
        int t   = idx >> 8;
        int col = (idx & 255) * 2;            // even col in [0,512)
        int j   = col & 31;                   // index within head dim
        float c0 = costab[t * DH + j],     s0 = sintab[t * DH + j];
        float c1 = costab[t * DH + j + 1], s1 = sintab[t * DH + j + 1];
        v2h* p  = (v2h*)(sQKV + t * QKVW + col);
        v2h  pv = *p;
        float e = (float)pv[0], o = (float)pv[1];
        v2h  rv;
        rv[0] = (_Float16)(e * c0 - o * s0);
        rv[1] = (_Float16)(o * c1 + e * s1);
        *p = rv;
    }
    __syncthreads();

    // ---- Phase 5: per-head attention; wave h owns head h ----
    {
        const int h = wid;
        const _Float16* qb = sQKV + h * DH;
        const _Float16* kb = sQKV + 256 + h * DH;
        const _Float16* vb = sQKV + 512 + h * DH;
        _Float16* atn = sXn + h * (64 * 64);          // per-wave 64x64 f16 scratch
        const float sscale = 0.17677669529663687f;    // 1/sqrt(32)

        v16h bK[4];
        #pragma unroll
        for (int ni = 0; ni < 4; ni++)
            bK[ni] = load_B_rows(kb, QKVW, ni * 16, 0, lane);
        v16h bV[2][2];
        #pragma unroll
        for (int kc = 0; kc < 2; kc++)
            #pragma unroll
            for (int nt = 0; nt < 2; nt++)
                bV[kc][nt] = load_B_strided(vb, QKVW, nt * 16, kc * 32, lane);

        for (int mi = 0; mi < 4; mi++) {
            v16h aQ = load_A(qb, QKVW, mi * 16, 0, lane);   // K = 32 = head dim
            v8f S[4];
            #pragma unroll
            for (int ni = 0; ni < 4; ni++) {
                v8f z = {};
                S[ni] = wmma_f16(aQ, bK[ni], z);
            }
            float rowmax[8], rowsum[8];
            #pragma unroll
            for (int r = 0; r < 8; r++) rowmax[r] = -3.0e38f;
            #pragma unroll
            for (int ni = 0; ni < 4; ni++) {
                bool inv = (ni == 3) && ((lane & 15) != 0);   // mask cols >= 49
                #pragma unroll
                for (int r = 0; r < 8; r++) {
                    float sv = inv ? -3.0e38f : S[ni][r] * sscale;
                    S[ni][r] = sv;
                    rowmax[r] = fmaxf(rowmax[r], sv);
                }
            }
            #pragma unroll
            for (int r = 0; r < 8; r++)
                #pragma unroll
                for (int m = 1; m < 16; m <<= 1)
                    rowmax[r] = fmaxf(rowmax[r], __shfl_xor(rowmax[r], m, 32));
            #pragma unroll
            for (int r = 0; r < 8; r++) rowsum[r] = 0.f;
            #pragma unroll
            for (int ni = 0; ni < 4; ni++)
                #pragma unroll
                for (int r = 0; r < 8; r++) {
                    float p = __expf(S[ni][r] - rowmax[r]);
                    S[ni][r] = p; rowsum[r] += p;
                }
            #pragma unroll
            for (int r = 0; r < 8; r++)
                #pragma unroll
                for (int m = 1; m < 16; m <<= 1)
                    rowsum[r] += __shfl_xor(rowsum[r], m, 32);
            int rbase = mi * 16 + 8 * (lane >> 4);
            int colb  = lane & 15;
            #pragma unroll
            for (int ni = 0; ni < 4; ni++)
                #pragma unroll
                for (int r = 0; r < 8; r++)
                    atn[(rbase + r) * 64 + ni * 16 + colb] =
                        (_Float16)(S[ni][r] / rowsum[r]);
        }

        // out = attn(64x64) @ v(64x32), gate, store into dead q-region as f16
        for (int mi = 0; mi < 4; mi++) {
            #pragma unroll
            for (int nt = 0; nt < 2; nt++) {
                v8f acc = {};
                #pragma unroll
                for (int kc = 0; kc < 2; kc++) {
                    v16h aA = load_A(atn, 64, mi * 16, kc * 32, lane);
                    acc = wmma_f16(aA, bV[kc][nt], acc);
                }
                int rbase = mi * 16 + 8 * (lane >> 4);
                int col   = h * DH + nt * 16 + (lane & 15);
                #pragma unroll
                for (int r = 0; r < 8; r++) {
                    int m = rbase + r;
                    float gv = (m < NTOK) ? sG[m * HEADS + h] : 0.f;
                    sQKV[m * QKVW + col] = (_Float16)(acc[r] * gv);
                }
            }
        }
    }
    __syncthreads();

    // ---- Phase 6: y = out_f(64x256) @ Wout -> global f32; A-fragments hoisted ----
    {
        const int mi = wid & 3;
        v16h aF[8];
        #pragma unroll
        for (int kk = 0; kk < 8; kk++)
            aF[kk] = load_A(sQKV, QKVW, mi * 16, kk * 32, lane);  // out_f in q-region
        for (int nt = (wid >> 2); nt < 32; nt += 2) {
            v8f acc = {};
            #pragma unroll
            for (int kk = 0; kk < 8; kk++) {
                v16h b = load_B_rows(woutT, 256, nt * 16, kk * 32, lane);
                acc = wmma_f16(aF[kk], b, acc);
            }
            int col   = nt * 16 + (lane & 15);
            int rbase = mi * 16 + 8 * (lane >> 4);
            #pragma unroll
            for (int r = 0; r < 8; r++) {
                int m = rbase + r;
                if (m < NTOK) ow[m * DIMX + col] = acc[r];
            }
        }
    }
}

extern "C" void kernel_launch(void* const* d_in, const int* in_sizes, int n_in,
                              void* d_out, int out_size, void* d_ws, size_t ws_size,
                              hipStream_t stream) {
    const float* x     = (const float*)d_in[0];
    const float* rot   = (const float*)d_in[1];
    const float* gamma = (const float*)d_in[2];
    const float* wqkv  = (const float*)d_in[3];
    const float* wg    = (const float*)d_in[4];
    const float* bg    = (const float*)d_in[5];
    const float* wout  = (const float*)d_in[6];
    float* out = (float*)d_out;

    char* ws = (char*)d_ws;
    size_t off = 0;
    _Float16* wqkvT  = (_Float16*)(ws + off); off += (size_t)QKVW * DIMX * 2;
    _Float16* woutT  = (_Float16*)(ws + off); off += (size_t)DIMX * 256 * 2;
    float*    costab = (float*)(ws + off);    off += (size_t)NTOK * DH * 4;
    float*    sintab = (float*)(ws + off);    off += (size_t)NTOK * DH * 4;
    _Float16* wgT    = (_Float16*)(ws + off); off += (size_t)16 * DIMX * 2;

    attn_prep<<<(QKVW * DIMX + 255) / 256, 256, 0, stream>>>(
        wqkv, wout, rot, wg, wqkvT, woutT, costab, sintab, wgT);

    const size_t smem = (size_t)(NPAD * DIMX + NPAD * QKVW) * 2 + NPAD * HEADS * 4;
    attn_main<<<2048, 256, smem, stream>>>(
        x, gamma, bg, wqkvT, woutT, costab, sintab, wgT, out);
}
